// GlobalEdgeGcnnLarge_44942537786158
// MI455X (gfx1250) — matrix operand order repaired and checked by reading, not compiled, not run on hardware
//
#include <hip/hip_runtime.h>

// Problem constants (fixed by the reference)
#define NN       100000
#define EE       400000
#define CC       16
#define LL       16
#define EPS_BN   1e-5f
#define NSLOPE   0.01f

typedef __attribute__((ext_vector_type(16))) _Float16 v16h;
typedef __attribute__((ext_vector_type(8)))  float    v8f;

// ---------------------------------------------------------------------------
// Utility: zero a float buffer
// ---------------------------------------------------------------------------
__global__ void k_zero(float* __restrict__ p, int n) {
    int i = blockIdx.x * blockDim.x + threadIdx.x;
    if (i < n) p[i] = 0.0f;
}

// ---------------------------------------------------------------------------
// Init: x0 = segment_sum(feats over src and dst); deg; deg_in (raw counts)
// One thread per (edge, channel) element.
// ---------------------------------------------------------------------------
__global__ void k_init_scatter(const float* __restrict__ ef,
                               const int* __restrict__ src,
                               const int* __restrict__ dst,
                               float* __restrict__ x0,
                               float* __restrict__ deg,
                               float* __restrict__ degin,
                               int total) {
    int i = blockIdx.x * blockDim.x + threadIdx.x;
    if (i >= total) return;
    int e = i >> 4;
    int c = i & 15;
    float f = ef[i];
    int s = src[e];
    int d = dst[e];
    atomicAdd(&x0[s * CC + c], f);
    atomicAdd(&x0[d * CC + c], f);
    if (c == 0) {
        atomicAdd(&deg[s], 1.0f);
        atomicAdd(&deg[d], 1.0f);
        atomicAdd(&degin[d], 1.0f);
    }
}

// x0 /= clip(deg,1);  degin = clip(degin,1)
__global__ void k_init_norm(float* __restrict__ x0,
                            const float* __restrict__ deg,
                            float* __restrict__ degin,
                            int total) {
    int i = blockIdx.x * blockDim.x + threadIdx.x;
    if (i >= total) return;
    int n = i >> 4;
    x0[i] /= fmaxf(deg[n], 1.0f);
    if ((i & 15) == 0) degin[n] = fmaxf(degin[n], 1.0f);
}

// ---------------------------------------------------------------------------
// Core: gather 16-edge tile [x[ia]|x[ib]] (16x32), f16-convert, WMMA with
// W^T (32x16), add bias, store h (E x 16) and accumulate per-channel
// sum / sumsq for BatchNorm (block LDS reduce -> 32 global atomics).
// 8 waves/block, 1 tile/wave. nTiles must keep full waves active (E%128==0).
// ---------------------------------------------------------------------------
__global__ void k_gemm_gather(const float* __restrict__ x,
                              const int* __restrict__ ia,
                              const int* __restrict__ ib,
                              const float* __restrict__ W,     // (16 x 32) row-major
                              const float* __restrict__ bias,  // (16)
                              float* __restrict__ out,         // (E x 16)
                              float* __restrict__ stats,       // [16 sum | 16 sumsq]
                              int nTiles) {
    __shared__ float shs[CC];
    __shared__ float shq[CC];
    const int lane = threadIdx.x & 31;
    const int wave = threadIdx.x >> 5;
    const int tile = blockIdx.x * 8 + wave;
    const int half = lane >> 4;   // 0 or 1
    const int nn   = lane & 15;   // output channel / row-in-tile selector

    if (threadIdx.x < CC) { shs[threadIdx.x] = 0.0f; shq[threadIdx.x] = 0.0f; }
    __syncthreads();

    if (tile < nTiles) {            // wave-uniform; EXEC stays all-ones inside
        // B operand: lane holds column nn, elems 0..7 = K half*8+e, elems 8..15 = K 16+half*8+e
        v16h b;
        const float* wr = W + nn * 32 + half * 8;
        #pragma unroll
        for (int e = 0; e < 8; ++e) { b[e] = (_Float16)wr[e]; b[8 + e] = (_Float16)wr[16 + e]; }

        // A operand: lane holds row (edge) nn of the tile.
        //   elems 0..7  = K half*8..half*8+7      -> x[ia] channels half*8..+7
        //   elems 8..15 = K 16+half*8..16+half*8+7 -> x[ib] channels half*8..+7
        const int edge = tile * CC + nn;
        const float* pa = x + (size_t)ia[edge] * CC + half * 8;
        const float* pb = x + (size_t)ib[edge] * CC + half * 8;
        v16h a;
        #pragma unroll
        for (int e = 0; e < 8; ++e) {
            a[e]     = (_Float16)pa[e];
            a[8 + e] = (_Float16)pb[e];
        }

        v8f c = {};
        c = __builtin_amdgcn_wmma_f32_16x16x32_f16(
                /*neg_a=*/false, a, /*neg_b=*/false, b,
                /*c_mod=*/(short)0, c, /*reuse_a=*/false, /*reuse_b=*/false);

        // D layout: lane group (half) holds rows half*8+v, column nn.
        const float bv = bias[nn];
        const int rowBase = tile * CC + half * 8;
        float s = 0.0f, q = 0.0f;
        #pragma unroll
        for (int v = 0; v < 8; ++v) {
            float val = c[v] + bv;
            out[(size_t)(rowBase + v) * CC + nn] = val;
            s += val;
            q += val * val;
        }
        atomicAdd(&shs[nn], s);   // ds_add_f32
        atomicAdd(&shq[nn], q);
    }
    __syncthreads();
    if (threadIdx.x < CC) {
        atomicAdd(&stats[threadIdx.x],      shs[threadIdx.x]);
        atomicAdd(&stats[CC + threadIdx.x], shq[threadIdx.x]);
    }
}

// ---------------------------------------------------------------------------
// BN finalize: scale = g * rsqrt(var+eps), shift = bt - mu*scale.
// Also resets the raw stats accumulator for the next use.
// ---------------------------------------------------------------------------
__global__ void k_bn_finalize(float* __restrict__ stats,
                              const float* __restrict__ g,
                              const float* __restrict__ bt,
                              float* __restrict__ scale,
                              float* __restrict__ shift,
                              float invCount) {
    int c = threadIdx.x;
    if (c < CC) {
        float mu  = stats[c] * invCount;
        float var = stats[CC + c] * invCount - mu * mu;
        float r   = rsqrtf(var + EPS_BN);
        float sc  = g[c] * r;
        scale[c] = sc;
        shift[c] = bt[c] - mu * sc;
        stats[c] = 0.0f;
        stats[CC + c] = 0.0f;
    }
}

// BN-affine + leaky-relu + scatter into x_next via atomics
__global__ void k_apply_scatter(const float* __restrict__ m,
                                const int* __restrict__ dst,
                                const float* __restrict__ scale,
                                const float* __restrict__ shift,
                                float* __restrict__ xnext,
                                int total) {
    int i = blockIdx.x * blockDim.x + threadIdx.x;
    if (i >= total) return;
    int e = i >> 4;
    int c = i & 15;
    float v = m[i] * scale[c] + shift[c];
    v = (v > 0.0f) ? v : NSLOPE * v;
    atomicAdd(&xnext[(size_t)dst[e] * CC + c], v);
}

// x_next /= deg_in (already clipped)
__global__ void k_node_norm(float* __restrict__ xn,
                            const float* __restrict__ degin,
                            int total) {
    int i = blockIdx.x * blockDim.x + threadIdx.x;
    if (i >= total) return;
    xn[i] /= degin[i >> 4];
}

// ---------------------------------------------------------------------------
// Final: e_fwd/e_bwd affine + leaky, edge_out = 0.5*(f+b),
// side_loss partial = sum((f-b)^2) reduced block-wise.
// ---------------------------------------------------------------------------
__global__ void k_final_apply(const float* __restrict__ hf,
                              const float* __restrict__ hb,
                              const float* __restrict__ scale,
                              const float* __restrict__ shift,
                              float* __restrict__ out,
                              float* __restrict__ loss,
                              int total) {
    __shared__ float red;
    if (threadIdx.x == 0) red = 0.0f;
    __syncthreads();
    int i = blockIdx.x * blockDim.x + threadIdx.x;
    float dsq = 0.0f;
    if (i < total) {
        int c = i & 15;
        float a = hf[i] * scale[c] + shift[c];
        a = (a > 0.0f) ? a : NSLOPE * a;
        float b = hb[i] * scale[c] + shift[c];
        b = (b > 0.0f) ? b : NSLOPE * b;
        out[i] = 0.5f * (a + b);
        float d = a - b;
        dsq = d * d;
    }
    // wave-level pre-reduce (wave32) to cut LDS atomics
    #pragma unroll
    for (int off = 16; off > 0; off >>= 1)
        dsq += __shfl_down(dsq, off, 32);
    if ((threadIdx.x & 31) == 0) atomicAdd(&red, dsq);
    __syncthreads();
    if (threadIdx.x == 0) atomicAdd(loss, red);
}

__global__ void k_loss_finalize(float* __restrict__ out_loss,
                                const float* __restrict__ loss,
                                float inv) {
    out_loss[0] = loss[0] * inv;
}

// ---------------------------------------------------------------------------
// Launcher
// ---------------------------------------------------------------------------
extern "C" void kernel_launch(void* const* d_in, const int* in_sizes, int n_in,
                              void* d_out, int out_size, void* d_ws, size_t ws_size,
                              hipStream_t stream) {
    const float* ef  = (const float*)d_in[0];   // (E,16)
    const int*   ei  = (const int*)  d_in[1];   // (2,E)
    // d_in[2] = angles (unused by reference forward)
    const float* Wn  = (const float*)d_in[3];   // (L,16,32)
    const float* bn  = (const float*)d_in[4];   // (L,16)
    const float* gn  = (const float*)d_in[5];   // (L,16)
    const float* btn = (const float*)d_in[6];   // (L,16)
    const float* We  = (const float*)d_in[7];   // (16,32)
    const float* be  = (const float*)d_in[8];   // (16)
    const float* ge  = (const float*)d_in[9];   // (16)
    const float* bte = (const float*)d_in[10];  // (16)

    const int* src = ei;
    const int* dst = ei + EE;

    // Workspace layout (floats)
    float* w = (float*)d_ws;
    float* xA    = w; w += (size_t)NN * CC;   // node feats (ping)
    float* xB    = w; w += (size_t)NN * CC;   // node feats (pong)
    float* deg   = w; w += NN;
    float* degin = w; w += NN;
    float* m     = w; w += (size_t)EE * CC;   // layer activations / h_fwd
    float* hb    = w; w += (size_t)EE * CC;   // h_bwd
    float* stats = w; w += 2 * CC;
    float* scale = w; w += CC;
    float* shift = w; w += CC;
    float* loss  = w; w += 1;

    const int NT   = NN * CC;                 // 1.6M
    const int ET   = EE * CC;                 // 6.4M
    const int TPB  = 256;
    const dim3 gN((NT + TPB - 1) / TPB);
    const dim3 gE((ET + TPB - 1) / TPB);
    const dim3 gN1((NN + TPB - 1) / TPB);
    const int nTiles = EE / CC;               // 25000
    const dim3 gGemm(nTiles / 8);             // 8 waves/block, 1 tile/wave

    // --- re-zero all accumulators (graph replay safe) ---
    k_zero<<<gN,  TPB, 0, stream>>>(xA, NT);
    k_zero<<<gN1, TPB, 0, stream>>>(deg, NN);
    k_zero<<<gN1, TPB, 0, stream>>>(degin, NN);
    k_zero<<<1, 64, 0, stream>>>(stats, 2 * CC);
    k_zero<<<1, 32, 0, stream>>>(loss, 1);

    // --- node init: x0 = mean of incident edge feats ---
    k_init_scatter<<<gE, TPB, 0, stream>>>(ef, src, dst, xA, deg, degin, ET);
    k_init_norm<<<gN, TPB, 0, stream>>>(xA, deg, degin, NT);

    // --- L GCN layers ---
    float* xcur = xA;
    float* xnxt = xB;
    for (int l = 0; l < LL; ++l) {
        // m = [x[dst]|x[src]] @ W_l^T + b_l   (+ BN stats)
        k_gemm_gather<<<gGemm, TPB, 0, stream>>>(
            xcur, dst, src, Wn + (size_t)l * CC * 2 * CC, bn + (size_t)l * CC,
            m, stats, nTiles);
        k_bn_finalize<<<1, CC, 0, stream>>>(
            stats, gn + (size_t)l * CC, btn + (size_t)l * CC,
            scale, shift, 1.0f / (float)EE);
        k_zero<<<gN, TPB, 0, stream>>>(xnxt, NT);
        k_apply_scatter<<<gE, TPB, 0, stream>>>(m, dst, scale, shift, xnxt, ET);
        k_node_norm<<<gN, TPB, 0, stream>>>(xnxt, degin, NT);
        float* t = xcur; xcur = xnxt; xnxt = t;
    }

    // --- final edge MLP on both orderings, BN over 2E rows ---
    k_gemm_gather<<<gGemm, TPB, 0, stream>>>(xcur, src, dst, We, be, m,  stats, nTiles);
    k_gemm_gather<<<gGemm, TPB, 0, stream>>>(xcur, dst, src, We, be, hb, stats, nTiles);
    k_bn_finalize<<<1, CC, 0, stream>>>(stats, ge, bte, scale, shift,
                                        1.0f / (2.0f * (float)EE));

    float* out = (float*)d_out;
    k_final_apply<<<gE, TPB, 0, stream>>>(m, hb, scale, shift, out, loss, ET);
    k_loss_finalize<<<1, 1, 0, stream>>>(out + ET, loss, 1.0f / (float)ET);
}
